// PaiNN_17514876634213
// MI455X (gfx1250) — matrix operand order repaired and checked by reading, not compiled, not run on hardware
//
#include <hip/hip_runtime.h>
#include <cstdint>
#include <cstddef>

typedef __bf16 bf16_t;
typedef __attribute__((ext_vector_type(16))) __bf16 v16bf;
typedef __attribute__((ext_vector_type(8)))  __bf16 v8bf;
typedef __attribute__((ext_vector_type(2)))  __bf16 v2bf;
typedef __attribute__((ext_vector_type(8)))  float  v8f;

#define BM 128
#define BN 128
#define BK 32
#define LDA 40   // padded LDS pitch: 80B = 20 dwords -> conflict-free 16-lane strides
#define LDB 40

// Fast SiLU: hardware v_rcp_f32 (~1 ulp) instead of IEEE division — plenty
// for a bf16 pipeline.
__device__ __forceinline__ float silu_f(float v) {
    return v * __builtin_amdgcn_rcpf(1.0f + __expf(-v));
}

// Load 8 consecutive elements as bf16 (wide, branch-free).
__device__ __forceinline__ v8bf load8_bf16(const float* p) {
    const float4 a = *(const float4*)p;
    const float4 b = *(const float4*)(p + 4);
    v8bf r;
    r[0] = (bf16_t)a.x; r[1] = (bf16_t)a.y; r[2] = (bf16_t)a.z; r[3] = (bf16_t)a.w;
    r[4] = (bf16_t)b.x; r[5] = (bf16_t)b.y; r[6] = (bf16_t)b.z; r[7] = (bf16_t)b.w;
    return r;
}
__device__ __forceinline__ v8bf load8_bf16(const bf16_t* p) {
    return *(const v8bf*)p;
}

// ---------------------------------------------------------------------------
// Tiled GEMM: C[M,N] = epilogue(A[M,K] @ W[K,N] [+ bias]), via v_wmma bf16.
//   ACT == 1 -> SiLU;  SCALE -> multiply row m by rowscale[m];  HB -> add bias.
// Block = 256 threads (8 waves); block tile 128x128; wave tile 32x64
// (2 A-frags x 4 B-frags -> 8 WMMAs per k-step, 8 f32 accumulators).
// REQUIRES: M % 128 == 0, N % 128 == 0, K % 32 == 0 (callers pre-pad K).
// ---------------------------------------------------------------------------
template <typename AT, typename OT, int ACT, bool SCALE, bool HB>
__global__ __launch_bounds__(256)
void gemm_bf16_wmma(const AT* __restrict__ A, const float* __restrict__ W,
                    const float* __restrict__ bias, OT* __restrict__ C,
                    int Mg, int Ng, int Kg,
                    const float* __restrict__ rowscale)
{
    __shared__ bf16_t As[BM * LDA];
    __shared__ bf16_t Bs[BN * LDB];

    const int tid  = threadIdx.x;
    const int lane = tid & 31;
    const int wave = tid >> 5;
    const int wm   = wave & 3;      // m strip: wm*32
    const int wn   = wave >> 2;     // n strip: wn*64
    const int mb   = blockIdx.x * BM;
    const int nb   = blockIdx.y * BN;
    const int kh   = lane >> 4;     // lane half (0/1) per ISA 16-bit layouts
    const int l16  = lane & 15;

    // staging coordinates (branch-free, fully covers the tiles)
    const int ar = tid >> 1;            // A row    0..127
    const int ac = (tid & 1) * 16;      // A col    {0,16}
    const int bn = tid & 127;           // B column 0..127
    const int bk = (tid >> 7) * 16;     // B k base {0,16}

    v8f acc[2][4] = {};

    for (int k0 = 0; k0 < Kg; k0 += BK) {
        // prefetch next K tile into cache (global_prefetch_b8)
        if (k0 + BK < Kg) {
            __builtin_prefetch(A + (size_t)(mb + ar) * Kg + (k0 + BK + ac), 0, 3);
            __builtin_prefetch(W + (size_t)(k0 + BK + bk) * Ng + (nb + bn), 0, 3);
        }
        __syncthreads();
        // ---- stage A tile (BM x BK): two 16B LDS stores per thread ----
        {
            const AT* ap = A + (size_t)(mb + ar) * Kg + (k0 + ac);
            *(v8bf*)&As[ar * LDA + ac]     = load8_bf16(ap);
            *(v8bf*)&As[ar * LDA + ac + 8] = load8_bf16(ap + 8);
        }
        // ---- stage B tile (BK x BN) transposed: 8 packed pair-stores ----
        {
            const float* wp = W + (size_t)(k0 + bk) * Ng + (nb + bn);
            bf16_t* bsp = &Bs[bn * LDB + bk];
            #pragma unroll
            for (int p = 0; p < 8; ++p) {
                const float v0 = wp[0];
                const float v1 = wp[Ng];
                wp += 2 * (size_t)Ng;
                v2bf pk;
                pk[0] = (bf16_t)v0;
                pk[1] = (bf16_t)v1;
                *(v2bf*)(bsp + 2 * p) = pk;
            }
        }
        __syncthreads();

        union F { v16bf v; v8bf h[2]; };
        // ---- 2 A fragments: rows m = wm*32 + t2*16 + l16 ----
        F af[2];
        #pragma unroll
        for (int t2 = 0; t2 < 2; ++t2) {
            const bf16_t* arow = &As[(wm * 32 + t2 * 16 + l16) * LDA];
            af[t2].h[0] = *(const v8bf*)(arow + kh * 8);
            af[t2].h[1] = *(const v8bf*)(arow + 16 + kh * 8);
        }
        // ---- 4 B fragments: cols n = wn*64 + s*16 + l16 ----
        F bfv[4];
        #pragma unroll
        for (int s = 0; s < 4; ++s) {
            const bf16_t* bcol = &Bs[(wn * 64 + s * 16 + l16) * LDB + kh * 8];
            bfv[s].h[0] = *(const v8bf*)(bcol);
            bfv[s].h[1] = *(const v8bf*)(bcol + 16);
        }
        // ---- 8 independent WMMAs ----
        #pragma unroll
        for (int t2 = 0; t2 < 2; ++t2) {
            #pragma unroll
            for (int s = 0; s < 4; ++s) {
                acc[t2][s] = __builtin_amdgcn_wmma_f32_16x16x32_bf16(
                                 false, af[t2].v, false, bfv[s].v, (short)0,
                                 acc[t2][s], false, false);
            }
        }
    }

    // ---- epilogue: C/D layout -> lane l16 = n, VGPR r -> m = kh*8 + r ----
    #pragma unroll
    for (int t2 = 0; t2 < 2; ++t2) {
        #pragma unroll
        for (int s = 0; s < 4; ++s) {
            const int nn = nb + wn * 64 + s * 16 + l16;
            const float bv = HB ? bias[nn] : 0.0f;
            #pragma unroll
            for (int r = 0; r < 8; ++r) {
                const int mm = mb + wm * 32 + t2 * 16 + kh * 8 + r;
                float v = acc[t2][s][r] + bv;
                if (ACT) v = silu_f(v);
                if (SCALE) v *= rowscale[mm];
                C[(size_t)mm * Ng + nn] = (OT)v;
            }
        }
    }
}

// ---------------------------------------------------------------------------
__global__ void zero_f32(float* __restrict__ p, long long n) {
    long long i = (long long)blockIdx.x * blockDim.x + threadIdx.x;
    if (i < n) p[i] = 0.0f;
}

// rbf [E,R] f32 -> rbf_pad [E,32] bf16 (zero-padded cols R..31)
__global__ void pad_rbf(const float* __restrict__ rbf, bf16_t* __restrict__ out,
                        long long E, int R) {
    long long i = (long long)blockIdx.x * blockDim.x + threadIdx.x;
    if (i >= E * 32) return;
    const long long e = i >> 5;
    const int c = (int)(i & 31);
    out[i] = (bf16_t)((c < R) ? rbf[e * R + c] : 0.0f);
}

// W [R,Ncols] f32 -> out [32,Ncols] f32 (zero-padded rows R..31)
__global__ void pad_w(const float* __restrict__ w, float* __restrict__ out,
                      int R, int Ncols) {
    int i = blockIdx.x * blockDim.x + threadIdx.x;
    if (i >= 32 * Ncols) return;
    const int k = i / Ncols;
    out[i] = (k < R) ? w[i] : 0.0f;   // row-major: i = k*Ncols + n
}

// Per-edge gather * filters, scatter-add into node messages. 1 edge / block,
// channel-per-lane (H = 128 = blockDim).
__global__ __launch_bounds__(128)
void edge_scatter(const int* __restrict__ ei, const bf16_t* __restrict__ filt,
                  const float* __restrict__ x, const float* __restrict__ mu,
                  const float* __restrict__ uv,
                  float* __restrict__ smsg, float* __restrict__ vmsg,
                  float* __restrict__ deg, int E)
{
    const int e = blockIdx.x;
    const int t = threadIdx.x;          // channel 0..127
    const int tgt = ei[e];
    const int src = ei[E + e];

    const size_t fb = (size_t)e * 384;
    const float fq = (float)filt[fb + t];
    const float fr = (float)filt[fb + 128 + t];
    const float fm = (float)filt[fb + 256 + t];

    const size_t xb = (size_t)src * 384;
    const float xq = x[xb + t]        * fq;
    const float xr = x[xb + 128 + t]  * fr;
    const float xm = x[xb + 256 + t]  * fm;

    atomicAdd(&smsg[(size_t)tgt * 128 + t], xq);

    const float u0 = uv[(size_t)e * 3 + 0];
    const float u1 = uv[(size_t)e * 3 + 1];
    const float u2 = uv[(size_t)e * 3 + 2];
    const size_t sb = (size_t)src * 384;
    const size_t vb = (size_t)tgt * 384;
    atomicAdd(&vmsg[vb + t],        u0 * xr + mu[sb + t]        * xm);
    atomicAdd(&vmsg[vb + 128 + t],  u1 * xr + mu[sb + 128 + t]  * xm);
    atomicAdd(&vmsg[vb + 256 + t],  u2 * xr + mu[sb + 256 + t]  * xm);
    if (t == 0) atomicAdd(&deg[tgt], 1.0f);
}

__global__ void node_update(const float* __restrict__ q, const float* __restrict__ mu,
                            const float* __restrict__ smsg, const float* __restrict__ vmsg,
                            const float* __restrict__ deg,
                            float* __restrict__ q1, float* __restrict__ mu1, int Nn)
{
    long long i = (long long)blockIdx.x * blockDim.x + threadIdx.x;
    if (i >= (long long)Nn * 128) return;
    const int n = (int)(i >> 7);
    const int h = (int)(i & 127);
    const float inv = 1.0f / fmaxf(deg[n], 1.0f);
    q1[i] = q[i] + smsg[i] * inv;
    const size_t b = (size_t)n * 384 + h;
    mu1[b]       = mu[b]       + vmsg[b]       * inv;
    mu1[b + 128] = mu[b + 128] + vmsg[b + 128] * inv;
    mu1[b + 256] = mu[b + 256] + vmsg[b + 256] * inv;
}

__global__ void mix_prep(const float* __restrict__ q1, const float* __restrict__ mucat,
                         bf16_t* __restrict__ sin_, float* __restrict__ inner, int Nn)
{
    long long i = (long long)blockIdx.x * blockDim.x + threadIdx.x;
    if (i >= (long long)Nn * 128) return;
    const int n = (int)(i >> 7);
    const int h = (int)(i & 127);
    const size_t b = (size_t)n * 768;            // 3 rows * 256 cols
    const float v0 = mucat[b + h],       w0 = mucat[b + 128 + h];
    const float v1 = mucat[b + 256 + h], w1 = mucat[b + 384 + h];
    const float v2 = mucat[b + 512 + h], w2 = mucat[b + 640 + h];
    inner[i] = v0 * w0 + v1 * w1 + v2 * w2;
    const float nrm = sqrtf(v0 * v0 + v1 * v1 + v2 * v2 + 1e-8f);
    sin_[(size_t)n * 256 + h]       = (bf16_t)q1[i];
    sin_[(size_t)n * 256 + 128 + h] = (bf16_t)nrm;
}

__global__ void final_update(const float* __restrict__ q1, const float* __restrict__ mu1,
                             const float* __restrict__ mucat, const float* __restrict__ delta,
                             const float* __restrict__ inner,
                             float* __restrict__ qout, float* __restrict__ muout, int Nn)
{
    long long i = (long long)blockIdx.x * blockDim.x + threadIdx.x;
    if (i >= (long long)Nn * 128) return;
    const int n = (int)(i >> 7);
    const int h = (int)(i & 127);
    const size_t db = (size_t)n * 384;
    const float dq   = delta[db + h];
    const float dms  = delta[db + 128 + h];
    const float dqmu = delta[db + 256 + h];
    qout[i] = q1[i] + dq + dqmu * inner[i];
    const size_t mb = (size_t)n * 384 + h;
    const size_t cb = (size_t)n * 768;
    muout[mb]       = mu1[mb]       + mucat[cb + 128 + h] * dms;
    muout[mb + 128] = mu1[mb + 128] + mucat[cb + 384 + h] * dms;
    muout[mb + 256] = mu1[mb + 256] + mucat[cb + 640 + h] * dms;
}

// ---------------------------------------------------------------------------
extern "C" void kernel_launch(void* const* d_in, const int* in_sizes, int n_in,
                              void* d_out, int out_size, void* d_ws, size_t ws_size,
                              hipStream_t stream)
{
    (void)n_in; (void)out_size; (void)ws_size;
    const int H  = 128;
    const int Nn = in_sizes[0] / H;          // 32768
    const int E  = in_sizes[5];              // 262144
    const int R  = in_sizes[3] / E;          // 20

    const float* q   = (const float*)d_in[0];
    const float* mu  = (const float*)d_in[1];
    const int*   ei  = (const int*)  d_in[2];
    const float* rbf = (const float*)d_in[3];
    const float* uv  = (const float*)d_in[4];
    const float* cut = (const float*)d_in[5];
    const float* Wi1 = (const float*)d_in[6];
    const float* bi1 = (const float*)d_in[7];
    const float* Wi2 = (const float*)d_in[8];
    const float* bi2 = (const float*)d_in[9];
    const float* Wf1 = (const float*)d_in[10];
    const float* bf1 = (const float*)d_in[11];
    const float* Wf2 = (const float*)d_in[12];
    const float* bf2 = (const float*)d_in[13];
    const float* Wv  = (const float*)d_in[14];
    const float* Ws1 = (const float*)d_in[15];
    const float* bs1 = (const float*)d_in[16];
    const float* Ws2 = (const float*)d_in[17];
    const float* bs2 = (const float*)d_in[18];

    // ---- workspace carving ----
    char* ws = (char*)d_ws;
    size_t off = 0;
    auto take = [&](size_t bytes) -> char* {
        char* p = ws + off;
        off = (off + bytes + 255) & ~(size_t)255;
        return p;
    };
    bf16_t* filt  = (bf16_t*)take((size_t)E * 384 * sizeof(bf16_t));   // 201 MB
    bf16_t* hf    = (bf16_t*)take((size_t)E * 128 * sizeof(bf16_t));   //  67 MB
    bf16_t* rbfp  = (bf16_t*)take((size_t)E * 32  * sizeof(bf16_t));   //  17 MB
    float*  Wf1p  = (float*) take((size_t)32 * H  * sizeof(float));
    bf16_t* hnod  = (bf16_t*)take((size_t)Nn * 384 * sizeof(bf16_t));
    float*  x     = (float*) take((size_t)Nn * 384 * sizeof(float));
    float*  smsg  = (float*) take((size_t)Nn * 513 * sizeof(float));   // smsg|vmsg|deg
    float*  vmsg  = smsg + (size_t)Nn * 128;
    float*  deg   = smsg + (size_t)Nn * 512;
    float*  q1    = (float*) take((size_t)Nn * 128 * sizeof(float));
    float*  mu1   = (float*) take((size_t)Nn * 384 * sizeof(float));
    bf16_t* sin_  = (bf16_t*)take((size_t)Nn * 256 * sizeof(bf16_t));
    float*  inner = (float*) take((size_t)Nn * 128 * sizeof(float));
    bf16_t* s1    = (bf16_t*)take((size_t)Nn * 384 * sizeof(bf16_t));
    float*  delta = (float*) take((size_t)Nn * 384 * sizeof(float));
    float*  mucat = (float*)filt;            // alias: filters dead before mixing

    float* qout  = (float*)d_out;
    float* muout = qout + (size_t)Nn * 128;

    const dim3 blk(256);

    // 0) zero-pad the K=20 operands to K=32 (GEMM requires K % 32 == 0)
    {
        long long tot = (long long)E * 32;
        pad_rbf<<<dim3((unsigned)((tot + 255) / 256)), blk, 0, stream>>>(rbf, rbfp, E, R);
        pad_w<<<dim3((32 * H + 255) / 256), blk, 0, stream>>>(Wf1, Wf1p, R, H);
    }

    // 1) hf = silu(rbf_pad @ Wf1_pad + bf1)    [E,32]@[32,128]
    gemm_bf16_wmma<bf16_t, bf16_t, 1, false, true>
        <<<dim3(E / BM, 128 / BN), blk, 0, stream>>>(
        rbfp, Wf1p, bf1, hf, E, 128, 32, nullptr);
    // 2) filters = (hf @ Wf2 + bf2) * cutoff   [E,128]@[128,384]  (dominant GEMM)
    gemm_bf16_wmma<bf16_t, bf16_t, 0, true, true>
        <<<dim3(E / BM, 384 / BN), blk, 0, stream>>>(
        hf, Wf2, bf2, filt, E, 384, 128, cut);
    // 3) h = silu(q @ Wi1 + bi1)               [N,128]@[128,384]
    gemm_bf16_wmma<float, bf16_t, 1, false, true>
        <<<dim3(Nn / BM, 384 / BN), blk, 0, stream>>>(
        q, Wi1, bi1, hnod, Nn, 384, 128, nullptr);
    // 4) x = h @ Wi2 + bi2                     [N,384]@[384,384]
    gemm_bf16_wmma<bf16_t, float, 0, false, true>
        <<<dim3(Nn / BM, 384 / BN), blk, 0, stream>>>(
        hnod, Wi2, bi2, x, Nn, 384, 384, nullptr);

    // 5) zero message accumulators, then edge gather/scatter
    {
        long long tot = (long long)Nn * 513;
        zero_f32<<<dim3((unsigned)((tot + 255) / 256)), blk, 0, stream>>>(smsg, tot);
    }
    edge_scatter<<<dim3(E), dim3(128), 0, stream>>>(ei, filt, x, mu, uv,
                                                    smsg, vmsg, deg, E);
    // 6) q1/mu1 = residual + msg/deg
    node_update<<<dim3((Nn * 128 + 255) / 256), blk, 0, stream>>>(
        q, mu, smsg, vmsg, deg, q1, mu1, Nn);

    // 7) mu_cat = mu1 @ Wv                     [3N,128]@[128,256]
    gemm_bf16_wmma<float, float, 0, false, false>
        <<<dim3((3 * Nn) / BM, 256 / BN), blk, 0, stream>>>(
        mu1, Wv, nullptr, mucat, 3 * Nn, 256, 128, nullptr);
    // 8) norms / inner / scalar_input
    mix_prep<<<dim3((Nn * 128 + 255) / 256), blk, 0, stream>>>(
        q1, mucat, sin_, inner, Nn);
    // 9) s1 = silu(scalar_input @ Ws1 + bs1)   [N,256]@[256,384]
    gemm_bf16_wmma<bf16_t, bf16_t, 1, false, true>
        <<<dim3(Nn / BM, 384 / BN), blk, 0, stream>>>(
        sin_, Ws1, bs1, s1, Nn, 384, 256, nullptr);
    // 10) delta = s1 @ Ws2 + bs2               [N,384]@[384,384]
    gemm_bf16_wmma<bf16_t, float, 0, false, true>
        <<<dim3(Nn / BM, 384 / BN), blk, 0, stream>>>(
        s1, Ws2, bs2, delta, Nn, 384, 384, nullptr);
    // 11) fused final update -> d_out (q then mu)
    final_update<<<dim3((Nn * 128 + 255) / 256), blk, 0, stream>>>(
        q1, mu1, mucat, delta, inner, qout, muout, Nn);
}
